// RW_GNN_55765855371264
// MI455X (gfx1250) — matrix operand order repaired and checked by reading, not compiled, not run on hardware
//
#include <hip/hip_runtime.h>

typedef __attribute__((ext_vector_type(16))) _Float16 v16h;
typedef __attribute__((ext_vector_type(8)))  float    v8f;

#define HH 16
#define SS 8
#define DD 64
#define GNUM 512
#define FOUT 384
#define PEN 128
#define NCLS 10

// ---------------------------------------------------------------- CDNA5 async global->LDS copy
// the LDS destination address comes from the lds_off VGPR; generic shared pointers carry the LDS
// offset in their low 32 bits (LDS aperture truncation, ISA 10.2).
__device__ __forceinline__ void async_ld128(unsigned lds_off, const void* gaddr) {
    asm volatile("global_load_async_to_lds_b128 %0, %1, off"
                 :: "v"(lds_off), "v"(gaddr)
                 : "memory");
}
__device__ __forceinline__ void wait_async0() {
    asm volatile("s_wait_asynccnt 0x0" ::: "memory");
}

// ---------------------------------------------------------------- utilities
__global__ void k_zero(float* __restrict__ p, long n) {
    long i = (long)blockIdx.x * blockDim.x + threadIdx.x;
    long stride = (long)gridDim.x * blockDim.x;
    for (; i < n; i += stride) p[i] = 0.f;
}

// ---------------------------------------------------------------- prep: A and z powers
__global__ void k_prep0(const float* __restrict__ adj_hidden,  // [H, 28]
                        const float* __restrict__ feat,        // [H,S,D] = 8192
                        float* __restrict__ Abuf,              // [H,S,S] = 1024
                        float* __restrict__ zbuf0,             // [128,64]
                        _Float16* __restrict__ z16_0) {
    int tid = threadIdx.x;
    for (int i = tid; i < HH * SS * SS; i += 256) Abuf[i] = 0.f;
    __syncthreads();
    for (int i = tid; i < HH * 28; i += 256) {
        int h = i / 28, q = i % 28;
        int a = 0, rem = q, rowlen = SS - 1;
        while (rem >= rowlen) { rem -= rowlen; rowlen--; a++; }
        int b = a + 1 + rem;
        float v = adj_hidden[i];
        v = v > 0.f ? v : 0.f;
        Abuf[h * 64 + a * 8 + b] = v;
        Abuf[h * 64 + b * 8 + a] = v;
    }
    for (int i = tid; i < HH * SS * DD; i += 256) {
        float v = feat[i];
        zbuf0[i] = v;
        z16_0[i] = (_Float16)v;
    }
}

// z_l = A @ z_{l-1}  per head  (one launch per level)
__global__ void k_propz(const float* __restrict__ Abuf,
                        const float* __restrict__ zprev,   // [128,64]
                        float* __restrict__ zout,
                        _Float16* __restrict__ z16out) {
    int i = blockIdx.x * 256 + threadIdx.x;
    if (i >= 8192) return;
    int d = i & 63, hs = i >> 6, h = hs >> 3, s = hs & 7;
    float acc = 0.f;
#pragma unroll
    for (int b = 0; b < SS; ++b)
        acc += Abuf[h * 64 + s * 8 + b] * zprev[(h * 8 + b) * 64 + d];
    zout[i] = acc;
    z16out[i] = (_Float16)acc;
}

// ---------------------------------------------------------------- degree / scatter
__global__ void k_deg(const int* __restrict__ row, float* __restrict__ deg, int E) {
    int e = blockIdx.x * 256 + threadIdx.x;
    if (e < E) atomicAdd(&deg[row[e]], 1.0f);
}

__global__ void k_dinv(const float* __restrict__ deg, float* __restrict__ dinv, int N) {
    int n = blockIdx.x * 256 + threadIdx.x;
    if (n < N) dinv[n] = 1.0f / (deg[n] + 1.0f);   // +1 = self loop
}

// agg[col] += dinv[row] * cur[row]   (edges only; self-loop folded into linear load)
__global__ void k_scatter(const int* __restrict__ row, const int* __restrict__ col,
                          const float* __restrict__ dinv,
                          const float* __restrict__ cur,   // [N,64]
                          float* __restrict__ agg, int E) {
    long idx = (long)blockIdx.x * 256 + threadIdx.x;
    long total = (long)E * 16;
    if (idx >= total) return;
    int e = (int)(idx >> 4);
    int d0 = (int)(idx & 15) * 4;
    int r = row[e], c = col[e];
    float s = dinv[r];
    const float4 v = *reinterpret_cast<const float4*>(cur + (long)r * 64 + d0);
    float* dst = agg + (long)c * 64 + d0;
    atomicAdd(dst + 0, s * v.x);
    atomicAdd(dst + 1, s * v.y);
    atomicAdd(dst + 2, s * v.z);
    atomicAdd(dst + 3, s * v.w);
}

// ---------------------------------------------------------------- WMMA linear: out[N,64] = act(in[N,K] @ W^T + b)
// optional composed input: in + dinv[n]*sl (self-loop fold)
template <int K, int ACT>
__global__ void __launch_bounds__(128)
k_linear(const float* __restrict__ in_f32,     // [N,K]
         const float* __restrict__ sl,         // [N,K] or null
         const float* __restrict__ dinv,       // [N]   or null
         const float* __restrict__ W,          // [64,K] row-major f32
         const float* __restrict__ bias,       // [64]
         float* __restrict__ out_f32,          // [N,64]
         _Float16* __restrict__ out_f16) {
    __shared__ _Float16 As[16][K];
    __shared__ _Float16 Ws[64][K];
    int tid = threadIdx.x;
    long base = (long)blockIdx.x * 16;

    for (int i = tid; i < 64 * K; i += 128)
        Ws[i / K][i % K] = (_Float16)W[i];
    for (int i = tid; i < 16 * K; i += 128) {
        int r = i / K, c = i % K;
        float v = in_f32[(base + r) * K + c];
        if (sl) v = v + dinv[base + r] * sl[(base + r) * K + c];
        As[r][c] = (_Float16)v;
    }
    __syncthreads();

    int wave = tid >> 5, lane = tid & 31;
    int m = lane & 15, half = lane >> 4;
    v8f acc = {};
#pragma unroll
    for (int k0 = 0; k0 < K; k0 += 32) {
        v16h a, b;
#pragma unroll
        for (int v = 0; v < 8; ++v) {
            int kb = k0 + half * 16 + v * 2;
            a[2 * v]     = As[m][kb];
            a[2 * v + 1] = As[m][kb + 1];
            int colw = wave * 16 + m;
            b[2 * v]     = Ws[colw][k0 + half * 16 + 2 * v];
            b[2 * v + 1] = Ws[colw][k0 + half * 16 + 2 * v + 1];
        }
        acc = __builtin_amdgcn_wmma_f32_16x16x32_f16(false, a, false, b, (short)0, acc,
                                                     false, false);
    }
#pragma unroll
    for (int r = 0; r < 8; ++r) {
        long rowg = base + half * 8 + r;
        int colg = wave * 16 + m;
        float v = acc[r] + bias[colg];
        if (ACT == 1) v = 1.0f / (1.0f + __expf(-v));
        out_f32[rowg * 64 + colg] = v;
        out_f16[rowg * 64 + colg] = (_Float16)v;
    }
}

// ---------------------------------------------------------------- step: T = cur @ z^T, val = zx*T (or T*T), segment-sum
__global__ void __launch_bounds__(128)
k_step(const _Float16* __restrict__ cur16,  // [N,64]
       const _Float16* __restrict__ z16,    // [128,64]
       const _Float16* __restrict__ zx_in,  // [N,128] (steps 1,2)
       _Float16* __restrict__ zx_out,       // [N,128] (step 0)
       float* __restrict__ acc,             // [G,128]
       const int* __restrict__ batch, int step0) {
    __shared__ _Float16 As[16][64];
    __shared__ _Float16 Zs[128][64];
    __shared__ float Tt[16][132];
    int tid = threadIdx.x;
    long base = (long)blockIdx.x * 16;

    // CDNA5 async DMA: global -> LDS without VGPR round-trip (ASYNCcnt)
    {
        unsigned lds_z = (unsigned)(uintptr_t)(&Zs[0][0]);
        unsigned lds_a = (unsigned)(uintptr_t)(&As[0][0]);
        const char* gz = (const char*)z16;
        const char* ga = (const char*)(cur16 + base * 64);
#pragma unroll
        for (int i = 0; i < 8; ++i) {   // Zs: 16 KB = 1024 x b128
            int chunk = tid + i * 128;
            async_ld128(lds_z + chunk * 16, gz + chunk * 16);
        }
        async_ld128(lds_a + tid * 16, ga + tid * 16);  // As: 2 KB = 128 x b128
        // prefetch the zx tile consumed after the barrier
        if (!step0) __builtin_prefetch((const char*)(zx_in + base * 128) + tid * 32, 0, 1);
        wait_async0();
    }
    __syncthreads();

    int wave = tid >> 5, lane = tid & 31;
    int m = lane & 15, half = lane >> 4;
#pragma unroll
    for (int ct = 0; ct < 2; ++ct) {
        int col0 = wave * 32 + ct * 16;
        v8f accv = {};
#pragma unroll
        for (int k0 = 0; k0 < 64; k0 += 32) {
            v16h a, b;
#pragma unroll
            for (int v = 0; v < 8; ++v) {
                int kb = k0 + half * 16 + v * 2;
                a[2 * v]     = As[m][kb];
                a[2 * v + 1] = As[m][kb + 1];
                b[2 * v]     = Zs[col0 + m][k0 + half * 16 + 2 * v];
                b[2 * v + 1] = Zs[col0 + m][k0 + half * 16 + 2 * v + 1];
            }
            accv = __builtin_amdgcn_wmma_f32_16x16x32_f16(false, a, false, b, (short)0,
                                                          accv, false, false);
        }
#pragma unroll
        for (int r = 0; r < 8; ++r) Tt[half * 8 + r][col0 + m] = accv[r];
    }
    __syncthreads();

    // fused elementwise + segmented reduce over sorted batch (atomics only on id change)
    int c = tid;  // 0..127
    float run = 0.f;
    int curg = batch[base];
    for (int rr = 0; rr < 16; ++rr) {
        long n = base + rr;
        float t = Tt[rr][c];
        float val;
        if (step0) {
            zx_out[n * 128 + c] = (_Float16)t;
            val = t * t;
        } else {
            val = (float)zx_in[n * 128 + c] * t;
        }
        int g = batch[n];
        if (g != curg) {
            atomicAdd(&acc[(long)curg * 128 + c], run);
            run = 0.f;
            curg = g;
        }
        run += val;
    }
    atomicAdd(&acc[(long)curg * 128 + c], run);
}

// ---------------------------------------------------------------- tail
// faithful reshape: flat = s*G*H + g*H + h ; out[flat>>7][i*128 + (flat&127)] = acc_i[g][h*8+s]
__global__ void k_assemble(const float* __restrict__ acc,  // [3][G][128]
                           float* __restrict__ outpre) {   // [G][384]
    int idx = blockIdx.x * 256 + threadIdx.x;
    if (idx >= 3 * SS * GNUM * HH) return;
    int i = idx / (SS * GNUM * HH);
    int f = idx % (SS * GNUM * HH);
    int s = f / (GNUM * HH);
    int g = (f % (GNUM * HH)) / HH;
    int h = f % HH;
    float v = acc[(long)i * GNUM * 128 + (long)g * 128 + h * 8 + s];
    outpre[(long)(f >> 7) * FOUT + i * 128 + (f & 127)] = v;
}

__global__ void k_bnstats(const float* __restrict__ outpre,
                          const float* __restrict__ gamma,
                          const float* __restrict__ beta,
                          float* __restrict__ bnp) {  // [2][384] scale, shift
    int j = threadIdx.x;
    if (j >= FOUT) return;
    float sum = 0.f, sumsq = 0.f;
    for (int g = 0; g < GNUM; ++g) {
        float v = outpre[(long)g * FOUT + j];
        sum += v;
        sumsq += v * v;
    }
    float mean = sum / GNUM;
    float var = sumsq / GNUM - mean * mean;
    float scale = gamma[j] * rsqrtf(var + 1e-5f);
    bnp[j] = scale;
    bnp[FOUT + j] = beta[j] - mean * scale;
}

__global__ void __launch_bounds__(128)
k_head(const float* __restrict__ outpre, const float* __restrict__ bnp,
       const float* __restrict__ fc1_w, const float* __restrict__ fc1_b,
       const float* __restrict__ fc2_w, const float* __restrict__ fc2_b,
       float* __restrict__ out) {  // [G,10]
    __shared__ float nrow[FOUT];
    __shared__ float h1[PEN];
    __shared__ float logits[NCLS];
    __shared__ float red;
    int g = blockIdx.x, tid = threadIdx.x;
    for (int j = tid; j < FOUT; j += 128)
        nrow[j] = outpre[(long)g * FOUT + j] * bnp[j] + bnp[FOUT + j];
    __syncthreads();
    {
        float s = fc1_b[tid];
        for (int j = 0; j < FOUT; ++j) s += nrow[j] * fc1_w[tid * FOUT + j];
        h1[tid] = s > 0.f ? s : 0.f;
    }
    __syncthreads();
    if (tid < NCLS) {
        float s = fc2_b[tid];
        for (int p = 0; p < PEN; ++p) s += h1[p] * fc2_w[tid * PEN + p];
        logits[tid] = s;
    }
    __syncthreads();
    if (tid == 0) {
        float mx = logits[0];
        for (int c = 1; c < NCLS; ++c) mx = fmaxf(mx, logits[c]);
        float se = 0.f;
        for (int c = 0; c < NCLS; ++c) se += __expf(logits[c] - mx);
        red = mx + __logf(se);
    }
    __syncthreads();
    if (tid < NCLS) out[(long)g * NCLS + tid] = logits[tid] - red;
}

// ---------------------------------------------------------------- launcher
extern "C" void kernel_launch(void* const* d_in, const int* in_sizes, int n_in,
                              void* d_out, int out_size, void* d_ws, size_t ws_size,
                              hipStream_t stream) {
    const float* x        = (const float*)d_in[0];
    const int*   ei       = (const int*)d_in[1];
    const int*   batch    = (const int*)d_in[2];
    const float* adj_h    = (const float*)d_in[3];
    const float* feat_h   = (const float*)d_in[4];
    const float* fc_w     = (const float*)d_in[5];
    const float* fc_b     = (const float*)d_in[6];
    const float* rw_w     = (const float*)d_in[7];
    const float* rw_b     = (const float*)d_in[8];
    const float* bn_gamma = (const float*)d_in[9];
    const float* bn_beta  = (const float*)d_in[10];
    const float* fc1_w    = (const float*)d_in[11];
    const float* fc1_b    = (const float*)d_in[12];
    const float* fc2_w    = (const float*)d_in[13];
    const float* fc2_b    = (const float*)d_in[14];
    float* out = (float*)d_out;

    const long N = in_sizes[0] / 128;
    const long E = in_sizes[1] / 2;
    const int* erow = ei;
    const int* ecol = ei + E;

    // ---- workspace carve (float units, 64-float aligned)
    float* ws = (float*)d_ws;
    size_t off = 0;
    auto alloc = [&](size_t nf) { size_t o = off; off += (nf + 63) & ~(size_t)63; return o; };
    size_t o_deg   = alloc(N);
    size_t o_dinv  = alloc(N);
    size_t o_xh32  = alloc(N * 64);
    size_t o_c132  = alloc(N * 64);
    size_t o_c232  = alloc(N * 64);
    size_t o_agga  = alloc(N * 64);
    size_t o_aggb  = alloc(N * 64);
    size_t o_acc   = alloc(3 * (size_t)GNUM * 128);
    size_t o_pre   = alloc((size_t)GNUM * FOUT);
    size_t o_bnp   = alloc(2 * FOUT);
    size_t o_A     = alloc(HH * SS * SS);
    size_t o_zbuf  = alloc(3 * 128 * 64);
    // halves (count in floats: nh halfs -> nh/2 floats)
    size_t o_xh16  = alloc(N * 64 / 2);
    size_t o_c116  = alloc(N * 64 / 2);
    size_t o_c216  = alloc(N * 64 / 2);
    size_t o_zx16  = alloc(N * 128 / 2);
    size_t o_z16   = alloc(3 * 128 * 64 / 2);

    float* deg  = ws + o_deg;
    float* dinv = ws + o_dinv;
    float* xh32 = ws + o_xh32;
    float* c132 = ws + o_c132;
    float* c232 = ws + o_c232;
    float* agga = ws + o_agga;
    float* aggb = ws + o_aggb;
    float* acc  = ws + o_acc;
    float* pre  = ws + o_pre;
    float* bnp  = ws + o_bnp;
    float* Abuf = ws + o_A;
    float* zbuf = ws + o_zbuf;
    _Float16* xh16 = (_Float16*)(ws + o_xh16);
    _Float16* c116 = (_Float16*)(ws + o_c116);
    _Float16* c216 = (_Float16*)(ws + o_c216);
    _Float16* zx16 = (_Float16*)(ws + o_zx16);
    _Float16* z16  = (_Float16*)(ws + o_z16);

    const int nblk16 = (int)(N / 16);  // N = 100000, exact

    // ---- zero accumulators (ws is poisoned; determinism across replays)
    k_zero<<<2048, 256, 0, stream>>>(deg, N);
    k_zero<<<8192, 256, 0, stream>>>(agga, N * 64);
    k_zero<<<8192, 256, 0, stream>>>(aggb, N * 64);
    k_zero<<<768, 256, 0, stream>>>(acc, 3 * (long)GNUM * 128);

    // ---- hidden graph prep
    k_prep0<<<1, 256, 0, stream>>>(adj_h, feat_h, Abuf, zbuf, z16);
    k_propz<<<32, 256, 0, stream>>>(Abuf, zbuf, zbuf + 8192, z16 + 8192);
    k_propz<<<32, 256, 0, stream>>>(Abuf, zbuf + 8192, zbuf + 16384, z16 + 16384);

    // ---- degrees
    k_deg<<<(int)((E + 255) / 256), 256, 0, stream>>>(erow, deg, (int)E);
    k_dinv<<<(int)((N + 255) / 256), 256, 0, stream>>>(deg, dinv, (int)N);

    // ---- xh = sigmoid(x @ fc_w^T + b)
    k_linear<128, 1><<<nblk16, 128, 0, stream>>>(x, nullptr, nullptr, fc_w, fc_b, xh32, xh16);

    // ---- step 0: T = zx, val = zx^2
    k_step<<<nblk16, 128, 0, stream>>>(xh16, z16, nullptr, zx16, acc, batch, 1);

    // ---- step 1: rwconv + linear, then einsum/mul/reduce
    k_scatter<<<(int)((E * 16 + 255) / 256), 256, 0, stream>>>(erow, ecol, dinv, xh32, agga, (int)E);
    k_linear<64, 0><<<nblk16, 128, 0, stream>>>(agga, xh32, dinv, rw_w, rw_b, c132, c116);
    k_step<<<nblk16, 128, 0, stream>>>(c116, z16 + 8192, zx16, nullptr,
                                       acc + (long)GNUM * 128, batch, 0);

    // ---- step 2
    k_scatter<<<(int)((E * 16 + 255) / 256), 256, 0, stream>>>(erow, ecol, dinv, c132, aggb, (int)E);
    k_linear<64, 0><<<nblk16, 128, 0, stream>>>(aggb, c132, dinv, rw_w + 64 * 64, rw_b + 64,
                                                c232, c216);
    k_step<<<nblk16, 128, 0, stream>>>(c216, z16 + 16384, zx16, nullptr,
                                       acc + 2L * GNUM * 128, batch, 0);

    // ---- tail: assemble, batchnorm, MLP head, log_softmax
    k_assemble<<<768, 256, 0, stream>>>(acc, pre);
    k_bnstats<<<1, FOUT, 0, stream>>>(pre, bn_gamma, bn_beta, bnp);
    k_head<<<GNUM, 128, 0, stream>>>(pre, bnp, fc1_w, fc1_b, fc2_w, fc2_b, out);
}
